// GPS_46883863003220
// MI455X (gfx1250) — compile-verified, matrix operand order checked by reading, not run on hardware
//
#include <hip/hip_runtime.h>

// ---------------------------------------------------------------------------
// GraphGPS forward for MI455X (gfx1250, wave32, WMMA).
// GEMMs run on v_wmma_f32_16x16x32_f16 (f32 accumulate) with fused
// bias/residual/ReLU epilogues. Operands are staged into LDS *pre-swizzled
// into the WMMA register layout* so each lane loads a whole fragment with two
// ds_load_b128. Main GEMMs use a 128x128 block tile (8 waves x 32x64 wave
// tile, 8 WMMAs per 12 b128 loads) with a double-buffered K loop; grid is
// ordered N-tiles-fastest so A re-reads hit the 192MB L2.
// Message passing uses f32 global atomics (dest fits in L2).
// ---------------------------------------------------------------------------

#define C_DIM     256
#define PE_DIM    64
#define NODE_FEAT 192   // C - PE
#define S_NODES   64
#define N_HEADS   4
#define L_LAYERS  6
#define BN_EPS    1e-5f

typedef __attribute__((ext_vector_type(16))) _Float16 v16h;
typedef __attribute__((ext_vector_type(8)))  float    v8f;
typedef __attribute__((ext_vector_type(2)))  _Float16 h2v;

// ---- WMMA 16-bit fragment swizzle (ISA 7.12.2) ------------------------------
// A 16x32 tile: lane = row + 16*hi(K), element e = (K&7) | ((K>>4)<<3).
// B 32x16 tile: lane = col + 16*hi(K), same e(K). Even K -> even e, and
// (K, K+1) -> (e, e+1) in the same lane, so a K-pair is one 32-bit LDS store
// and a full 16-half fragment is 32 contiguous bytes per lane.
__device__ __forceinline__ int swz_hi(int c) { return (c >> 3) & 1; }
__device__ __forceinline__ int swz_e0(int c) { return (c & 7) | ((c >> 4) << 3); }

#define WMMA_F16(Af, Bf, Cacc)                                             \
  __builtin_amdgcn_wmma_f32_16x16x32_f16(false, (Af), false, (Bf),         \
                                         (short)0, (Cacc), false, false)

// ---------------------------------------------------------------------------
// Big tiled GEMM: out[M,NT] = act(A[M,K] @ W[K,NT] + bias [+ resid]).
// Block = 256 threads = 8 waves; block tile 128x128; wave tile 32x64
// (2 A-frags x 4 B-frags -> 8 WMMAs per K-step). Double-buffered LDS.
// Requires M % 128 == 0, NT % 128 == 0, K % 32 == 0.
// grid = (NT/128, M/128): N-tiles fastest -> A rows stay hot in L2.
// ---------------------------------------------------------------------------
template <bool RELU>
__global__ __launch_bounds__(256, 2)
void gemm_wmma_big(const float* __restrict__ A, const float* __restrict__ W,
                   const float* __restrict__ bias, const float* __restrict__ resid,
                   float* __restrict__ out, int M, int K, int NT) {
  __shared__ __align__(32) _Float16 saf[2][8][32][16];  // [buf][rowtile][lane][e]
  __shared__ __align__(32) _Float16 sbf[2][8][32][16];  // [buf][coltile][lane][e]
  const int tid  = threadIdx.x;
  const int lane = tid & 31;
  const int wave = tid >> 5;
  const int wr   = wave & 3;   // 4 row groups of 32 rows (2 rowtiles)
  const int wc   = wave >> 2;  // 2 col groups of 64 cols (4 coltiles)
  const int m0   = blockIdx.y * 128;
  const int n0   = blockIdx.x * 128;

  auto stage = [&](int buf, int k0) {
    // A tile: 128 rows x 16 K-pairs (float2 read -> one b32 swizzled store)
#pragma unroll
    for (int i = 0; i < 8; ++i) {
      const int e = tid + i * 256;  // 0..2047
      const int r = e >> 4, c = (e & 15) * 2;
      const float* ap = &A[(size_t)(m0 + r) * K + (k0 + c)];
      h2v v;
      v.x = (_Float16)ap[0];
      v.y = (_Float16)ap[1];
      *(h2v*)&saf[buf][r >> 4][(r & 15) + 16 * swz_hi(c)][swz_e0(c)] = v;
    }
    // W tile: 16 K-pairs x 128 cols (two coalesced row reads -> one b32 store)
#pragma unroll
    for (int i = 0; i < 8; ++i) {
      const int e = tid + i * 256;
      const int n = e & 127, k = (e >> 7) * 2;
      const float* wp = &W[(size_t)(k0 + k) * NT + (n0 + n)];
      h2v v;
      v.x = (_Float16)wp[0];
      v.y = (_Float16)wp[NT];
      *(h2v*)&sbf[buf][n >> 4][(n & 15) + 16 * swz_hi(k)][swz_e0(k)] = v;
    }
  };

  v8f acc[2][4];
#pragma unroll
  for (int rt = 0; rt < 2; ++rt)
#pragma unroll
    for (int bt = 0; bt < 4; ++bt) acc[rt][bt] = (v8f){};

  stage(0, 0);
  int buf = 0;
  for (int k0 = 0; k0 < K; k0 += 32) {
    __syncthreads();
    if (k0 + 32 < K) {
      if (k0 + 64 < K)  // pull K+2 A-slice toward L2 (global_prefetch_b8)
        __builtin_prefetch(&A[(size_t)(m0 + (tid >> 1)) * K + (k0 + 64)], 0, 0);
      stage(buf ^ 1, k0 + 32);  // overlaps with WMMAs below (other buffer)
    }
    const v16h a0 = *(const v16h*)&saf[buf][wr * 2][lane][0];      // 2x ds_load_b128
    const v16h a1 = *(const v16h*)&saf[buf][wr * 2 + 1][lane][0];
#pragma unroll
    for (int bt = 0; bt < 4; ++bt) {
      const v16h bf = *(const v16h*)&sbf[buf][wc * 4 + bt][lane][0];
      acc[0][bt] = WMMA_F16(a0, bf, acc[0][bt]);
      acc[1][bt] = WMMA_F16(a1, bf, acc[1][bt]);
    }
    buf ^= 1;
  }

  // D layout: VGPR r -> M = r (lanes 0-15) / r+8 (lanes 16-31); N = lane%16.
#pragma unroll
  for (int rt = 0; rt < 2; ++rt) {
    const int rbase = m0 + wr * 32 + rt * 16 + ((lane < 16) ? 0 : 8);
#pragma unroll
    for (int bt = 0; bt < 4; ++bt) {
      const int cc = n0 + wc * 64 + bt * 16 + (lane & 15);
      const float bb = bias[cc];
#pragma unroll
      for (int r = 0; r < 8; ++r) {
        const int rr = rbase + r;
        float v = acc[rt][bt][r] + bb;
        if (resid) v += resid[(size_t)rr * NT + cc];
        if (RELU) v = fmaxf(v, 0.f);
        out[(size_t)rr * NT + cc] = v;
      }
    }
  }
}

// ---------------------------------------------------------------------------
// Small tiled GEMM (64x64 block tile) for the NT=64 head projection.
// grid = (NT/64, M/64).
// ---------------------------------------------------------------------------
template <bool RELU>
__global__ __launch_bounds__(256, 2)
void gemm_wmma(const float* __restrict__ A, const float* __restrict__ W,
               const float* __restrict__ bias, const float* __restrict__ resid,
               float* __restrict__ out, int M, int K, int NT) {
  __shared__ __align__(32) _Float16 saf[2][4][32][16];
  __shared__ __align__(32) _Float16 sbf[2][4][32][16];
  const int tid  = threadIdx.x;
  const int lane = tid & 31;
  const int wave = tid >> 5;
  const int wr   = wave & 3;
  const int wc   = wave >> 2;
  const int m0   = blockIdx.y * 64;
  const int n0   = blockIdx.x * 64;

  auto stage = [&](int buf, int k0) {
#pragma unroll
    for (int i = 0; i < 4; ++i) {
      const int e = tid + i * 256;
      const int r = e >> 4, c = (e & 15) * 2;
      const float* ap = &A[(size_t)(m0 + r) * K + (k0 + c)];
      h2v v;
      v.x = (_Float16)ap[0];
      v.y = (_Float16)ap[1];
      *(h2v*)&saf[buf][r >> 4][(r & 15) + 16 * swz_hi(c)][swz_e0(c)] = v;
    }
#pragma unroll
    for (int i = 0; i < 4; ++i) {
      const int e = tid + i * 256;
      const int n = e & 63, k = (e >> 6) * 2;
      const float* wp = &W[(size_t)(k0 + k) * NT + (n0 + n)];
      h2v v;
      v.x = (_Float16)wp[0];
      v.y = (_Float16)wp[NT];
      *(h2v*)&sbf[buf][n >> 4][(n & 15) + 16 * swz_hi(k)][swz_e0(k)] = v;
    }
  };

  v8f acc0 = {}, acc1 = {};
  stage(0, 0);
  int buf = 0;
  for (int k0 = 0; k0 < K; k0 += 32) {
    __syncthreads();
    if (k0 + 32 < K) stage(buf ^ 1, k0 + 32);
    const v16h af = *(const v16h*)&saf[buf][wr][lane][0];
    const v16h b0 = *(const v16h*)&sbf[buf][wc * 2][lane][0];
    const v16h b1 = *(const v16h*)&sbf[buf][wc * 2 + 1][lane][0];
    acc0 = WMMA_F16(af, b0, acc0);
    acc1 = WMMA_F16(af, b1, acc1);
    buf ^= 1;
  }

  const int rbase = m0 + wr * 16 + ((lane < 16) ? 0 : 8);
  const int c0    = n0 + wc * 32 + (lane & 15);
  const int c1    = c0 + 16;
  const float bb0 = bias[c0], bb1 = bias[c1];
#pragma unroll
  for (int r = 0; r < 8; ++r) {
    const int rr = rbase + r;
    float v0 = acc0[r] + bb0;
    float v1 = acc1[r] + bb1;
    if (resid) {
      v0 += resid[(size_t)rr * NT + c0];
      v1 += resid[(size_t)rr * NT + c1];
    }
    if (RELU) { v0 = fmaxf(v0, 0.f); v1 = fmaxf(v1, 0.f); }
    out[(size_t)rr * NT + c0] = v0;
    out[(size_t)rr * NT + c1] = v1;
  }
}

// ---------------------------------------------------------------------------
// Attention: one block (4 waves) per (graph, head). S=64, d=64.
// scores = (q*scale) @ k^T via WMMA; softmax; o = P @ v via WMMA.
// q/kT/v/P all live in LDS in pre-swizzled fragment layout.
// ---------------------------------------------------------------------------
__global__ __launch_bounds__(128, 2)
void attn_wmma(const float* __restrict__ qkv, float* __restrict__ o) {
  const int g  = blockIdx.x >> 2;
  const int hh = blockIdx.x & 3;
  __shared__ __align__(32) _Float16 sqf[4][2][32][16];  // A frags of q [rowtile][ktile]
  __shared__ __align__(32) _Float16 skf[2][4][32][16];  // B frags of k^T [ktile][ntile]
  __shared__ __align__(32) _Float16 svf[2][4][32][16];  // B frags of v  [ktile][ntile]
  __shared__ __align__(32) _Float16 spf[4][2][32][16];  // A frags of P
  __shared__ float ssc[64][68];                         // f32 scores
  const int tid = threadIdx.x, lane = tid & 31, wave = tid >> 5;
  const float scale = 0.125f;  // 1/sqrt(64)

  const float* base = qkv + (size_t)g * S_NODES * 768 + hh * 64;
  // stage q (A layout, scaled) and k^T (B layout, K dim = d, col = node j)
  for (int i = tid; i < 64 * 32; i += 128) {
    const int s = i >> 5, d = (i & 31) * 2;
    const float* row = base + (size_t)s * 768;
    h2v vq;
    vq.x = (_Float16)(row[d] * scale);
    vq.y = (_Float16)(row[d + 1] * scale);
    *(h2v*)&sqf[s >> 4][d >> 5][(s & 15) + 16 * swz_hi(d & 31)][swz_e0(d & 31)] = vq;
    h2v vk;
    vk.x = (_Float16)row[256 + d];
    vk.y = (_Float16)row[256 + d + 1];
    *(h2v*)&skf[d >> 5][s >> 4][(s & 15) + 16 * swz_hi(d & 31)][swz_e0(d & 31)] = vk;
  }
  // stage v (B layout, K dim = node j, col = d)
  for (int i = tid; i < 64 * 32; i += 128) {
    const int j = (i >> 6) * 2, d = i & 63;
    const float* r0 = base + (size_t)j * 768 + 512;
    h2v vv;
    vv.x = (_Float16)r0[d];
    vv.y = (_Float16)r0[768 + d];
    *(h2v*)&svf[j >> 5][d >> 4][(d & 15) + 16 * swz_hi(j & 31)][swz_e0(j & 31)] = vv;
  }
  __syncthreads();

  const int col = lane & 15;
  const int rb  = wave * 16 + ((lane < 16) ? 0 : 8);

  // scores
#pragma unroll
  for (int nt = 0; nt < 4; ++nt) {
    v8f acc = {};
#pragma unroll
    for (int kt = 0; kt < 2; ++kt) {
      const v16h af = *(const v16h*)&sqf[wave][kt][lane][0];
      const v16h bf = *(const v16h*)&skf[kt][nt][lane][0];
      acc = WMMA_F16(af, bf, acc);
    }
#pragma unroll
    for (int r = 0; r < 8; ++r) ssc[rb + r][nt * 16 + col] = acc[r];
  }
  __syncthreads();

  // row softmax; write probs directly in A-fragment layout
  if (tid < 64) {
    float mx = -1e30f;
    for (int j = 0; j < 64; ++j) mx = fmaxf(mx, ssc[tid][j]);
    float sum = 0.f;
    for (int j = 0; j < 64; ++j) {
      const float e = __expf(ssc[tid][j] - mx);
      ssc[tid][j] = e;
      sum += e;
    }
    const float inv = 1.f / sum;
    for (int jp = 0; jp < 32; ++jp) {
      const int j = jp * 2;
      h2v vp;
      vp.x = (_Float16)(ssc[tid][j] * inv);
      vp.y = (_Float16)(ssc[tid][j + 1] * inv);
      *(h2v*)&spf[tid >> 4][j >> 5][(tid & 15) + 16 * swz_hi(j & 31)][swz_e0(j & 31)] = vp;
    }
  }
  __syncthreads();

  // o = P @ V
#pragma unroll
  for (int nt = 0; nt < 4; ++nt) {
    v8f acc = {};
#pragma unroll
    for (int kt = 0; kt < 2; ++kt) {
      const v16h af = *(const v16h*)&spf[wave][kt][lane][0];
      const v16h bf = *(const v16h*)&svf[kt][nt][lane][0];
      acc = WMMA_F16(af, bf, acc);
    }
#pragma unroll
    for (int r = 0; r < 8; ++r) {
      const int n = g * 64 + rb + r;
      o[(size_t)n * C_DIM + hh * 64 + nt * 16 + col] = acc[r];
    }
  }
}

// ---------------------------------------------------------------------------
// Elementwise / reduction helpers
// ---------------------------------------------------------------------------
__global__ void zero_kernel(float* __restrict__ p, int n) {
  const int i = blockIdx.x * blockDim.x + threadIdx.x;
  if (i < n) p[i] = 0.f;
}

__global__ void copy4_kernel(const float4* __restrict__ s, float4* __restrict__ d,
                             size_t n4) {
  const size_t i = blockIdx.x * (size_t)blockDim.x + threadIdx.x;
  if (i < n4) d[i] = s[i];
}

__global__ void add_kernel(const float* __restrict__ a, const float* __restrict__ b,
                           float* __restrict__ o, size_t n) {
  const size_t i = blockIdx.x * (size_t)blockDim.x + threadIdx.x;
  if (i < n) o[i] = a[i] + b[i];
}

// column sums + sums of squares (coalesced: threads span columns)
__global__ __launch_bounds__(256)
void colsum_kernel(const float* __restrict__ X, int Nrows, int K,
                   float* __restrict__ sum, float* __restrict__ sumsq) {
  const int rpb = (Nrows + gridDim.x - 1) / gridDim.x;
  const int r0 = blockIdx.x * rpb;
  const int r1 = min(r0 + rpb, Nrows);
  for (int c = threadIdx.x; c < K; c += blockDim.x) {
    float s = 0.f, s2 = 0.f;
    for (int r = r0; r < r1; ++r) {
      const float v = X[(size_t)r * K + c];
      s += v;
      s2 += v * v;
    }
    atomicAdd(&sum[c], s);
    atomicAdd(&sumsq[c], s2);
  }
}

__global__ void bnfin_kernel(const float* __restrict__ sum,
                             const float* __restrict__ sumsq, int Nrows, int K,
                             float* __restrict__ mean, float* __restrict__ rstd) {
  const int c = blockIdx.x * blockDim.x + threadIdx.x;
  if (c < K) {
    const float m = sum[c] / (float)Nrows;
    const float v = sumsq[c] / (float)Nrows - m * m;
    mean[c] = m;
    rstd[c] = rsqrtf(v + BN_EPS);
  }
}

// K is always 256 here (power of two) -> cheap column extraction
__global__ void bnapply_kernel(const float* __restrict__ X, float* __restrict__ Y,
                               const float* __restrict__ mean,
                               const float* __restrict__ rstd,
                               const float* __restrict__ g,
                               const float* __restrict__ b, size_t n) {
  const size_t i = blockIdx.x * (size_t)blockDim.x + threadIdx.x;
  if (i < n) {
    const int c = (int)(i & (size_t)(C_DIM - 1));
    Y[i] = (X[i] - mean[c]) * rstd[c] * g[c] + b[c];
  }
}

// h = concat(node_emb[x], bn(pe) @ pe_lin_w + pe_lin_b)   (block = node, 256 thr)
__global__ __launch_bounds__(256)
void embed_kernel(const int* __restrict__ x, const float* __restrict__ pe,
                  const float* __restrict__ pmean, const float* __restrict__ prstd,
                  const float* __restrict__ pg, const float* __restrict__ pb,
                  const float* __restrict__ node_emb,
                  const float* __restrict__ pw, const float* __restrict__ pbias,
                  float* __restrict__ h) {
  const int i = blockIdx.x, c = threadIdx.x;
  if (c < NODE_FEAT) {
    h[(size_t)i * C_DIM + c] = node_emb[x[i] * NODE_FEAT + c];
  } else {
    const int oc = c - NODE_FEAT;
    float acc = pbias[oc];
#pragma unroll
    for (int k = 0; k < 20; ++k) {
      const float pn = (pe[(size_t)i * 20 + k] - pmean[k]) * prstd[k] * pg[k] + pb[k];
      acc += pn * pw[k * PE_DIM + oc];
    }
    h[(size_t)i * C_DIM + c] = acc;
  }
}

// z[dst] += relu(h[src] + edge_emb[attr])  (block = edge, 256 channels)
__global__ __launch_bounds__(256)
void gine_scatter(const float* __restrict__ h, const float* __restrict__ edge_emb,
                  const int* __restrict__ src, const int* __restrict__ dst,
                  const int* __restrict__ eattr, float* __restrict__ z) {
  const int e = blockIdx.x, c = threadIdx.x;
  const int s = src[e], d = dst[e], a = eattr[e];
  const float m = fmaxf(h[(size_t)s * C_DIM + c] + edge_emb[a * C_DIM + c], 0.f);
  atomicAdd(&z[(size_t)d * C_DIM + c], m);
}

// pooled[g,c] = sum over 64 nodes
__global__ __launch_bounds__(256)
void pool_kernel(const float* __restrict__ h, float* __restrict__ pooled) {
  const int g = blockIdx.x, c = threadIdx.x;
  float s = 0.f;
  for (int j = 0; j < S_NODES; ++j)
    s += h[((size_t)g * S_NODES + j) * C_DIM + c];
  pooled[(size_t)g * C_DIM + c] = s;
}

// final 64 -> 1 projection
__global__ void head3_kernel(const float* __restrict__ t, const float* __restrict__ w,
                             const float* __restrict__ b, float* __restrict__ out,
                             int G) {
  const int g = blockIdx.x * blockDim.x + threadIdx.x;
  if (g < G) {
    float s = b[0];
#pragma unroll
    for (int k = 0; k < 64; ++k) s += t[(size_t)g * 64 + k] * w[k];
    out[g] = s;
  }
}

// ---------------------------------------------------------------------------
// Host-side orchestration
// ---------------------------------------------------------------------------
static void run_bn(const float* X, float* Y, const float* g, const float* b,
                   int Nrows, float* ssum, float* ssq, float* smean,
                   float* srstd, hipStream_t stream) {
  zero_kernel<<<2, 256, 0, stream>>>(ssum, 512);  // ssum + ssq contiguous
  colsum_kernel<<<256, 256, 0, stream>>>(X, Nrows, C_DIM, ssum, ssq);
  bnfin_kernel<<<1, 256, 0, stream>>>(ssum, ssq, Nrows, C_DIM, smean, srstd);
  const size_t n = (size_t)Nrows * C_DIM;
  bnapply_kernel<<<(unsigned)((n + 255) / 256), 256, 0, stream>>>(X, Y, smean,
                                                                  srstd, g, b, n);
}

extern "C" void kernel_launch(void* const* d_in, const int* in_sizes, int n_in,
                              void* d_out, int out_size, void* d_ws, size_t ws_size,
                              hipStream_t stream) {
  const int*   x         = (const int*)d_in[0];
  const float* pe        = (const float*)d_in[1];
  const int*   ei        = (const int*)d_in[2];
  const int*   eattr     = (const int*)d_in[3];
  const float* node_emb  = (const float*)d_in[5];
  const float* pe_lin_w  = (const float*)d_in[6];
  const float* pe_lin_b  = (const float*)d_in[7];
  const float* pe_norm_g = (const float*)d_in[8];
  const float* pe_norm_b = (const float*)d_in[9];
  const float* edge_emb  = (const float*)d_in[10];
  const float* conv_w1   = (const float*)d_in[11];
  const float* conv_b1   = (const float*)d_in[12];
  const float* conv_w2   = (const float*)d_in[13];
  const float* conv_b2   = (const float*)d_in[14];
  const float* attn_in_w = (const float*)d_in[15];
  const float* attn_in_b = (const float*)d_in[16];
  const float* attn_o_w  = (const float*)d_in[17];
  const float* attn_o_b  = (const float*)d_in[18];
  const float* bn1_g     = (const float*)d_in[19];
  const float* bn1_b     = (const float*)d_in[20];
  const float* bn2_g     = (const float*)d_in[21];
  const float* bn2_b     = (const float*)d_in[22];
  const float* bn3_g     = (const float*)d_in[23];
  const float* bn3_b     = (const float*)d_in[24];
  const float* mlp_w1    = (const float*)d_in[25];
  const float* mlp_b1    = (const float*)d_in[26];
  const float* mlp_w2    = (const float*)d_in[27];
  const float* mlp_b2    = (const float*)d_in[28];
  const float* head_w1   = (const float*)d_in[29];
  const float* head_b1   = (const float*)d_in[30];
  const float* head_w2   = (const float*)d_in[31];
  const float* head_b2   = (const float*)d_in[32];
  const float* head_w3   = (const float*)d_in[33];
  const float* head_b3   = (const float*)d_in[34];

  const int N = in_sizes[0];   // 131072
  const int E = in_sizes[3];   // 524288
  const int G = N / S_NODES;   // 2048
  const int* src = ei;
  const int* dst = ei + E;

  // workspace layout (floats)
  float* ws = (float*)d_ws;
  const size_t NC = (size_t)N * C_DIM;
  float* h      = ws;
  float* z      = ws + NC;
  float* t1     = ws + 2 * NC;
  float* hl     = ws + 3 * NC;
  float* ha     = ws + 4 * NC;
  float* big    = ws + 5 * NC;               // N x 768 (qkv / FFN hidden)
  float* pooled = ws + 8 * NC;               // G x 256
  float* t2     = pooled + (size_t)G * 256;  // G x 128
  float* t3     = t2 + (size_t)G * 128;      // G x 64
  float* stats  = t3 + (size_t)G * 64;
  float* ssum = stats, *ssq = stats + 256, *smean = stats + 512, *srstd = stats + 768;

  const dim3 blk256(256), blk128(128);
  // grid = (N-tiles, M-tiles): column tiles of the same A rows run adjacently.
  const dim3 gB2(2, N / 128), gB4(4, N / 128), gB6(6, N / 128);

  // ---- input embedding: BN(pe) folded into embed ----
  zero_kernel<<<2, 256, 0, stream>>>(ssum, 512);
  colsum_kernel<<<256, 256, 0, stream>>>(pe, N, 20, ssum, ssq);
  bnfin_kernel<<<1, 256, 0, stream>>>(ssum, ssq, N, 20, smean, srstd);
  embed_kernel<<<N, 256, 0, stream>>>(x, pe, smean, srstd, pe_norm_g, pe_norm_b,
                                      node_emb, pe_lin_w, pe_lin_b, h);

  for (int l = 0; l < L_LAYERS; ++l) {
    const size_t WCC = (size_t)C_DIM * C_DIM;
    // ---- GINEConv: z = h + scatter_add(relu(h[src] + ea)) ----
    copy4_kernel<<<(unsigned)(NC / 4 / 256), 256, 0, stream>>>(
        (const float4*)h, (float4*)z, NC / 4);
    gine_scatter<<<E, 256, 0, stream>>>(h, edge_emb, src, dst, eattr, z);
    gemm_wmma_big<true><<<gB2, blk256, 0, stream>>>(z, conv_w1 + l * WCC,
        conv_b1 + l * C_DIM, nullptr, t1, N, C_DIM, C_DIM);
    gemm_wmma_big<false><<<gB2, blk256, 0, stream>>>(t1, conv_w2 + l * WCC,
        conv_b2 + l * C_DIM, h, hl, N, C_DIM, C_DIM);  // + residual h
    run_bn(hl, hl, bn1_g + l * C_DIM, bn1_b + l * C_DIM, N,
           ssum, ssq, smean, srstd, stream);

    // ---- global attention ----
    gemm_wmma_big<false><<<gB6, blk256, 0, stream>>>(h, attn_in_w + l * WCC * 3,
        attn_in_b + l * 3 * C_DIM, nullptr, big, N, C_DIM, 3 * C_DIM);
    attn_wmma<<<G * N_HEADS, blk128, 0, stream>>>(big, t1);
    gemm_wmma_big<false><<<gB2, blk256, 0, stream>>>(t1, attn_o_w + l * WCC,
        attn_o_b + l * C_DIM, h, ha, N, C_DIM, C_DIM);  // + residual h
    run_bn(ha, ha, bn2_g + l * C_DIM, bn2_b + l * C_DIM, N,
           ssum, ssq, smean, srstd, stream);

    // ---- combine + FFN ----
    add_kernel<<<(unsigned)(NC / 256), 256, 0, stream>>>(hl, ha, z, NC);
    gemm_wmma_big<true><<<gB4, blk256, 0, stream>>>(z, mlp_w1 + l * WCC * 2,
        mlp_b1 + l * 2 * C_DIM, nullptr, big, N, C_DIM, 2 * C_DIM);
    gemm_wmma_big<false><<<gB2, blk256, 0, stream>>>(big, mlp_w2 + l * WCC * 2,
        mlp_b2 + l * C_DIM, z, t1, N, 2 * C_DIM, C_DIM);  // + residual z
    run_bn(t1, h, bn3_g + l * C_DIM, bn3_b + l * C_DIM, N,
           ssum, ssq, smean, srstd, stream);
  }

  // ---- pool + prediction head ----
  pool_kernel<<<G, 256, 0, stream>>>(h, pooled);
  gemm_wmma_big<true><<<dim3(1, G / 128), blk256, 0, stream>>>(pooled, head_w1,
      head_b1, nullptr, t2, G, 256, 128);
  gemm_wmma<true><<<dim3(1, G / 64), blk256, 0, stream>>>(t2, head_w2,
      head_b2, nullptr, t3, G, 128, 64);
  head3_kernel<<<(G + 255) / 256, 256, 0, stream>>>(t3, head_w3, head_b3,
      (float*)d_out, G);
}
